// MultiTimescaleSNN_65747359367904
// MI455X (gfx1250) — matrix operand order repaired and checked by reading, not compiled, not run on hardware
//
#include <hip/hip_runtime.h>
#include <hip/hip_fp16.h>

// MultiTimescale SNN forward for MI455X (gfx1250).
//
// Strategy (see analysis): the workload is recurrence-latency bound, not
// compute bound (86 GFLOP of GEMM, 403 MB of mandatory output writes
// ~17us at 23.3 TB/s).  One persistent kernel, 16 workgroups, each owning
// 32 of the 512 hidden columns for the WHOLE batch so BatchNorm statistics
// are WG-local.  f16 WMMA (16x16x32, f32 accum) for all hidden GEMMs;
// membrane potentials live in VGPRs across all 256 timesteps in exactly the
// WMMA C-tile layout.  Only the binary spike matrices cross workgroups,
// through L2-resident workspace, guarded by a grid barrier
// (atomic counter + s_sleep spin, plus the CDNA5 cluster barrier which is a
// hardware barrier when cluster-dispatched and an architectural NOP otherwise).

typedef __attribute__((ext_vector_type(16))) _Float16 v16h;
typedef __attribute__((ext_vector_type(8)))  _Float16 v8h;
typedef __attribute__((ext_vector_type(8)))  float    v8f;

#define NWG    16
#define NTHR   256
#define TSTEPS 256
#define BATCH  256
#define DIN    256
#define HDIM   512
#define DOUT   3
#define BN_EPS 1e-5f

// ---------------------------------------------------------------------------
__global__ void snn_init(unsigned* ctr) {
  if (threadIdx.x == 0 && blockIdx.x == 0) *ctr = 0u;
}

// Grid-wide barrier: release fence -> atomic arrive -> spin (thread 0 only,
// others parked in s_barrier) -> cluster barrier (s_barrier_signal/wait -3;
// NOP if not cluster-dispatched) -> acquire fence.
__device__ __forceinline__ void grid_barrier(unsigned* ctr, unsigned target) {
  __syncthreads();
  if (threadIdx.x == 0) {
    __threadfence();  // make our spike/ws stores visible device-wide
    __hip_atomic_fetch_add(ctr, 1u, __ATOMIC_ACQ_REL, __HIP_MEMORY_SCOPE_AGENT);
    while (__hip_atomic_load(ctr, __ATOMIC_ACQUIRE, __HIP_MEMORY_SCOPE_AGENT) < target)
      __builtin_amdgcn_s_sleep(1);
  }
  __builtin_amdgcn_s_cluster_barrier();   // CDNA5 cluster barrier (-3)
  __syncthreads();
  __threadfence();  // acquire: don't read stale spike data
}

// ---------------------------------------------------------------------------
// One SNN layer: Y = Sin @ W  (B=256 x KD  @  KD x 32-cols-of-this-WG),
// then BN over the batch (per column), then LIF update + spike emit.
// Wave `wave` owns batch rows [wave*32, wave*32+32): M-tiles m0, m0+1.
// mem[4]: persistent membrane C-tiles [mt*2+nt] for this wave.
template<int KD>
__device__ __forceinline__ void snn_layer(
    const _Float16* __restrict__ Sin,   // [BATCH][KD]  spikes (f16, row major)
    const _Float16* __restrict__ Wt,    // [HDIM][KD]   W^T    (f16, K contiguous)
    const float* __restrict__ gamma, const float* __restrict__ beta_bn,
    float leak, v8f* mem,
    float* __restrict__ trace,          // d_out membrane trace slice [BATCH][HDIM]
    _Float16* __restrict__ Sout,        // [BATCH][HDIM] spikes out (f16)
    float* lds_psum, float* lds_psq, float* lds_mean, float* lds_rstd,
    int wg, int wave, int lane, int tid)
{
  const int half = lane >> 4;          // 0: lanes 0-15, 1: lanes 16-31
  const int lrow = lane & 15;
  const int m0   = wave * 2;           // first of this wave's two M-tiles

  v8f acc[4] = {};                     // C tiles [mt*2+nt]

  // Row/col base pointers for fragment loads (ISA 7.12.2 VGPR layouts):
  //  A 16x32 f16: lane -> row = lrow (+mtile*16); K halves {base..base+7,
  //               base+16..base+23} with base = half*8  -> two 16B loads.
  //  B 32x16 f16: lane -> col = lrow (+ntile*16); same K halves as A
  //               -> two 16B loads (Wt is [N][K]).
  const _Float16* arow0 = Sin + (size_t)(m0 * 16 + lrow) * KD;
  const _Float16* arow1 = arow0 + (size_t)16 * KD;
  const _Float16* bcol0 = Wt + (size_t)(wg * 32 + lrow) * KD;
  const _Float16* bcol1 = bcol0 + (size_t)16 * KD;

  for (int k0 = 0; k0 < KD; k0 += 32) {
    union { v16h v; v8h h[2]; } a0, a1;
    a0.h[0] = *(const v8h*)(arow0 + k0 + half * 8);
    a0.h[1] = *(const v8h*)(arow0 + k0 + 16 + half * 8);
    a1.h[0] = *(const v8h*)(arow1 + k0 + half * 8);
    a1.h[1] = *(const v8h*)(arow1 + k0 + 16 + half * 8);
    union { v16h v; v8h h[2]; } b0u, b1u;
    b0u.h[0] = *(const v8h*)(bcol0 + k0 + half * 8);
    b0u.h[1] = *(const v8h*)(bcol0 + k0 + 16 + half * 8);
    b1u.h[0] = *(const v8h*)(bcol1 + k0 + half * 8);
    b1u.h[1] = *(const v8h*)(bcol1 + k0 + 16 + half * 8);
    v16h b0 = b0u.v;
    v16h b1 = b1u.v;
    acc[0] = __builtin_amdgcn_wmma_f32_16x16x32_f16(false, a0.v, false, b0, (short)0, acc[0], false, false);
    acc[1] = __builtin_amdgcn_wmma_f32_16x16x32_f16(false, a0.v, false, b1, (short)0, acc[1], false, false);
    acc[2] = __builtin_amdgcn_wmma_f32_16x16x32_f16(false, a1.v, false, b0, (short)0, acc[2], false, false);
    acc[3] = __builtin_amdgcn_wmma_f32_16x16x32_f16(false, a1.v, false, b1, (short)0, acc[3], false, false);
  }

  // --- BatchNorm stats (per column, over all 256 batch rows). ------------
  // C layout: element v of acc tile = (row = mtile*16 + half*8 + v,
  // col = ntile*16 + lrow).  Per-lane partials cover 16 rows of one column;
  // lane L and L^16 hold the same column -> shfl_xor(16) combines to the
  // wave's 32 rows, then an 8-wave LDS reduction finishes the batch.
  float s0 = 0.f, q0 = 0.f, s1 = 0.f, q1 = 0.f;
#pragma unroll
  for (int v = 0; v < 8; ++v) {
    s0 += acc[0][v] + acc[2][v];
    q0 += acc[0][v] * acc[0][v] + acc[2][v] * acc[2][v];
    s1 += acc[1][v] + acc[3][v];
    q1 += acc[1][v] * acc[1][v] + acc[3][v] * acc[3][v];
  }
  s0 += __shfl_xor(s0, 16, 32);  q0 += __shfl_xor(q0, 16, 32);
  s1 += __shfl_xor(s1, 16, 32);  q1 += __shfl_xor(q1, 16, 32);
  if (lane < 16) {
    lds_psum[wave * 32 + lrow]      = s0;  lds_psq[wave * 32 + lrow]      = q0;
    lds_psum[wave * 32 + 16 + lrow] = s1;  lds_psq[wave * 32 + 16 + lrow] = q1;
  }
  __syncthreads();
  if (tid < 32) {
    float s = 0.f, q = 0.f;
#pragma unroll
    for (int w = 0; w < 8; ++w) { s += lds_psum[w * 32 + tid]; q += lds_psq[w * 32 + tid]; }
    const float mu  = s * (1.0f / BATCH);
    const float var = q * (1.0f / BATCH) - mu * mu;   // biased, as torch BN
    lds_mean[tid] = mu;
    lds_rstd[tid] = rsqrtf(var + BN_EPS);
  }
  __syncthreads();

  // --- BN apply + Leaky integrate-and-fire, all in registers. -------------
  // NOTE: fc bias cancels exactly through BN (shifts mu identically) -> skipped.
#pragma unroll
  for (int mt = 0; mt < 2; ++mt) {
#pragma unroll
    for (int nt = 0; nt < 2; ++nt) {
      const int ai  = mt * 2 + nt;
      const int c   = nt * 16 + lrow;
      const int col = wg * 32 + c;
      const float mu = lds_mean[c];
      const float rs = lds_rstd[c];
      const float gg = gamma[col];
      const float bv = beta_bn[col];
      const int rowb = (m0 + mt) * 16 + half * 8;
#pragma unroll
      for (int v = 0; v < 8; ++v) {
        const float xn = gg * (acc[ai][v] - mu) * rs + bv;
        const float mo = mem[ai][v];
        const float mn = leak * mo + xn - (mo > 1.0f ? 1.0f : 0.0f); // reset from PREV mem
        mem[ai][v] = mn;
        const int row = rowb + v;
        trace[row * HDIM + col] = mn;                           // membrane trace out
        Sout[row * HDIM + col]  = (_Float16)(mn > 1.0f ? 1.0f : 0.0f); // fire
      }
    }
  }
}

// ---------------------------------------------------------------------------
__global__ __launch_bounds__(NTHR) void snn_main(
    const float* __restrict__ spikes,
    const float* __restrict__ W0, const float* __restrict__ g0, const float* __restrict__ bb0,
    const float* __restrict__ W1, const float* __restrict__ g1, const float* __restrict__ bb1,
    const float* __restrict__ W2, const float* __restrict__ g2, const float* __restrict__ bb2,
    const float* __restrict__ Wo, const float* __restrict__ bo,
    unsigned* __restrict__ ctr,
    _Float16* __restrict__ SA, _Float16* __restrict__ S1,
    _Float16* __restrict__ S2, _Float16* __restrict__ S3,
    _Float16* __restrict__ Wt0, _Float16* __restrict__ Wt1, _Float16* __restrict__ Wt2,
    float* __restrict__ out)
{
  __shared__ float lds_psum[8 * 32];
  __shared__ float lds_psq[8 * 32];
  __shared__ float lds_mean[32];
  __shared__ float lds_rstd[32];

  const int tid  = threadIdx.x;
  const int wg   = blockIdx.x;
  const int wave = tid >> 5;
  const int lane = tid & 31;
  const int gtid = wg * NTHR + tid;
  unsigned bar = 0;

  // Prologue: transpose+convert weights to f16 [N][K] once (cooperative).
  for (int i = gtid; i < HDIM * DIN;  i += NWG * NTHR) { int n = i >> 8, k = i & (DIN  - 1); Wt0[i] = (_Float16)W0[k * HDIM + n]; }
  for (int i = gtid; i < HDIM * HDIM; i += NWG * NTHR) { int n = i >> 9, k = i & (HDIM - 1); Wt1[i] = (_Float16)W1[k * HDIM + n]; }
  for (int i = gtid; i < HDIM * HDIM; i += NWG * NTHR) { int n = i >> 9, k = i & (HDIM - 1); Wt2[i] = (_Float16)W2[k * HDIM + n]; }

  // Persistent membrane state for this wave's C-tiles: 3 layers x 4 tiles.
  v8f mem[3][4] = {};

  float* tr0 = out + (size_t)TSTEPS * BATCH * DOUT;
  float* tr1 = tr0 + (size_t)TSTEPS * BATCH * HDIM;
  float* tr2 = tr1 + (size_t)TSTEPS * BATCH * HDIM;

  grid_barrier(ctr, ++bar * NWG);   // weights ready

  for (int t = 0; t < TSTEPS; ++t) {
    // Convert this timestep's input spikes to f16; prefetch next timestep.
    const float* xin = spikes + (size_t)t * BATCH * DIN;
    for (int i = gtid; i < BATCH * DIN; i += NWG * NTHR) {
      SA[i] = (_Float16)xin[i];
      if (t + 1 < TSTEPS) __builtin_prefetch(xin + BATCH * DIN + i, 0, 0);
    }
    grid_barrier(ctr, ++bar * NWG);

    snn_layer<DIN >(SA, Wt0, g0, bb0, 0.90f, mem[0], tr0 + (size_t)t * BATCH * HDIM, S1,
                    lds_psum, lds_psq, lds_mean, lds_rstd, wg, wave, lane, tid);
    grid_barrier(ctr, ++bar * NWG);
    snn_layer<HDIM>(S1, Wt1, g1, bb1, 0.95f, mem[1], tr1 + (size_t)t * BATCH * HDIM, S2,
                    lds_psum, lds_psq, lds_mean, lds_rstd, wg, wave, lane, tid);
    grid_barrier(ctr, ++bar * NWG);
    snn_layer<HDIM>(S2, Wt2, g2, bb2, 0.99f, mem[2], tr2 + (size_t)t * BATCH * HDIM, S3,
                    lds_psum, lds_psq, lds_mean, lds_rstd, wg, wave, lane, tid);
    grid_barrier(ctr, ++bar * NWG);

    // Readout: out[t] = spk3 @ Wo + bo.  WG w owns batch rows [w*16, w*16+16).
    if (tid < 16 * DOUT) {
      const int r = wg * 16 + tid / DOUT;
      const int o = tid % DOUT;
      float acc = bo[o];
      for (int k = 0; k < HDIM; ++k) acc += (float)S3[r * HDIM + k] * Wo[k * DOUT + o];
      out[(size_t)t * BATCH * DOUT + r * DOUT + o] = acc;
    }
    // No barrier needed here: SA/S3 reuse hazards are covered by the next
    // iteration's barriers (all WGs must arrive at barrier A(t+1) before any
    // WG can reach the phase that overwrites S3, etc.).
  }
}

// ---------------------------------------------------------------------------
extern "C" void kernel_launch(void* const* d_in, const int* in_sizes, int n_in,
                              void* d_out, int out_size, void* d_ws, size_t ws_size,
                              hipStream_t stream) {
  (void)in_sizes; (void)n_in; (void)out_size; (void)ws_size;

  // setup_inputs order:
  // 0:spike_input 1:W0 2:b0 3:g0 4:bb0 5:W1 6:b1 7:g1 8:bb1
  // 9:W2 10:b2 11:g2 12:bb2 13:Wo 14:bo   (b0..b2 cancel through BN)
  const float* spikes = (const float*)d_in[0];
  const float* W0 = (const float*)d_in[1];
  const float* g0 = (const float*)d_in[3];
  const float* bb0 = (const float*)d_in[4];
  const float* W1 = (const float*)d_in[5];
  const float* g1 = (const float*)d_in[7];
  const float* bb1 = (const float*)d_in[8];
  const float* W2 = (const float*)d_in[9];
  const float* g2 = (const float*)d_in[11];
  const float* bb2 = (const float*)d_in[12];
  const float* Wo = (const float*)d_in[13];
  const float* bo = (const float*)d_in[14];
  float* out = (float*)d_out;

  // Workspace layout (~2.3 MB): barrier counter + f16 spike matrices + f16 W^T.
  char* ws = (char*)d_ws;
  size_t off = 0;
  unsigned* ctr = (unsigned*)(ws + off); off += 256;
  _Float16* SA  = (_Float16*)(ws + off); off += (size_t)BATCH * DIN  * 2;
  _Float16* S1  = (_Float16*)(ws + off); off += (size_t)BATCH * HDIM * 2;
  _Float16* S2  = (_Float16*)(ws + off); off += (size_t)BATCH * HDIM * 2;
  _Float16* S3  = (_Float16*)(ws + off); off += (size_t)BATCH * HDIM * 2;
  _Float16* Wt0 = (_Float16*)(ws + off); off += (size_t)HDIM * DIN  * 2;
  _Float16* Wt1 = (_Float16*)(ws + off); off += (size_t)HDIM * HDIM * 2;
  _Float16* Wt2 = (_Float16*)(ws + off); off += (size_t)HDIM * HDIM * 2;

  snn_init<<<1, 64, 0, stream>>>(ctr);   // ws is poisoned; re-zero barrier ctr
  snn_main<<<dim3(NWG), dim3(NTHR), 0, stream>>>(
      spikes, W0, g0, bb0, W1, g1, bb1, W2, g2, bb2, Wo, bo,
      ctr, SA, S1, S2, S3, Wt0, Wt1, Wt2, out);
}
